// GAT_35167192219921
// MI455X (gfx1250) — matrix operand order repaired and checked by reading, not compiled, not run on hardware
//
#include <hip/hip_runtime.h>
#include <hip/hip_bf16.h>
#include <math.h>

// ---------------------------------------------------------------------------
// GNN (GCN + 2xGAT + GraphNorm) for MI455X / gfx1250.
// GEMMs: bf16 WMMA (v_wmma_f32_16x16x32_bf16), W staged in LDS (padded, bank-
// conflict free b128 loads). Message passing: wave-per-edge float4 gather +
// f32 L2 atomics (feature buffers are 51MB -> resident in the 192MB L2).
// GraphNorm: run-length segment reduction over sorted batch ids.
// ---------------------------------------------------------------------------

#define NN   100000
#define EE   1600000
#define GGR  64
#define FF   128
#define HH   4
#define CCH  32
#define FOUT 64
#define LDSK 136   // padded bf16 K-stride: lanes 4 banks apart -> no DS conflicts

typedef __attribute__((ext_vector_type(16))) __bf16 v16bf;
typedef __attribute__((ext_vector_type(8)))  float  v8f;

union V16 { v16bf v; unsigned short u[16]; uint4 q[2]; };

__device__ __forceinline__ unsigned short f2bf(float f) {   // f32 -> bf16 RNE
  unsigned u = __float_as_uint(f);
  return (unsigned short)((u + 0x7FFFu + ((u >> 16) & 1u)) >> 16);
}
__device__ __forceinline__ unsigned encf(float f) {         // order-preserving f32->u32
  unsigned u = __float_as_uint(f);
  return (u & 0x80000000u) ? ~u : (u | 0x80000000u);
}
__device__ __forceinline__ float decf(unsigned u) {
  return (u & 0x80000000u) ? __uint_as_float(u & 0x7fffffffu) : __uint_as_float(~u);
}
__device__ __forceinline__ float leaky(float v) { return v > 0.f ? v : 0.2f * v; }

// ---------------- GEMM: out[M,ncols] = A[M,128] @ W[ncols,128]^T (+bias) ----
__global__ __launch_bounds__(256)
void k_gemm(const float* __restrict__ A, const float* __restrict__ W,
            const float* __restrict__ bias, float* __restrict__ out,
            int M, int ncols)
{
  __shared__ unsigned short sW[FF * LDSK];          // bf16 bits, padded stride
  const int tid = threadIdx.x;
  for (int idx = tid; idx < ncols * FF; idx += 256) {
    int f = idx >> 7, k = idx & 127;
    sW[f * LDSK + k] = f2bf(W[idx]);
  }
  __syncthreads();

  const int lane = tid & 31, wave = tid >> 5;
  const int hi = lane >> 4, rl = lane & 15;
  const int rowbase = blockIdx.x * 128 + wave * 16;
  int arow = rowbase + rl; if (arow >= M) arow = M - 1;   // clamp loads only

  // A fragments: 16-bit A 16x32 layout -> lane-lo K {0..7,16..23}, lane-hi +8
  V16 a[4];
  #pragma unroll
  for (int ks = 0; ks < 4; ++ks) {
    const float* ap = A + (long)arow * FF + ks * 32 + hi * 8;
    float4 p0 = *(const float4*)(ap);
    float4 p1 = *(const float4*)(ap + 4);
    float4 p2 = *(const float4*)(ap + 16);
    float4 p3 = *(const float4*)(ap + 20);
    a[ks].u[0]=f2bf(p0.x); a[ks].u[1]=f2bf(p0.y); a[ks].u[2]=f2bf(p0.z); a[ks].u[3]=f2bf(p0.w);
    a[ks].u[4]=f2bf(p1.x); a[ks].u[5]=f2bf(p1.y); a[ks].u[6]=f2bf(p1.z); a[ks].u[7]=f2bf(p1.w);
    a[ks].u[8]=f2bf(p2.x); a[ks].u[9]=f2bf(p2.y); a[ks].u[10]=f2bf(p2.z); a[ks].u[11]=f2bf(p2.w);
    a[ks].u[12]=f2bf(p3.x); a[ks].u[13]=f2bf(p3.y); a[ks].u[14]=f2bf(p3.z); a[ks].u[15]=f2bf(p3.w);
  }

  const int nct = ncols >> 4;
  for (int ct = 0; ct < nct; ++ct) {
    const int col = ct * 16 + rl;
    v8f acc = {};
    #pragma unroll
    for (int ks = 0; ks < 4; ++ks) {
      V16 b;   // B 32x16 layout: lane-lo K {0..7,16..23}, lane-hi +8 (matches A)
      const uint4* bp = (const uint4*)(&sW[col * LDSK + ks * 32 + hi * 8]);
      b.q[0] = bp[0];
      b.q[1] = bp[2];
      acc = __builtin_amdgcn_wmma_f32_16x16x32_bf16(false, a[ks].v, false, b.v,
                                                    (short)0, acc, false, false);
    }
    const float bv = bias ? bias[col] : 0.f;
    #pragma unroll
    for (int r = 0; r < 8; ++r) {                   // C/D: row = r + 8*laneHi
      int orow = rowbase + r + hi * 8;
      if (orow < M) out[(long)orow * ncols + col] = acc[r] + bv;
    }
  }
}

// ---------------- utility kernels ------------------------------------------
__global__ void k_fill(float* p, float v, long long n) {
  long long i = (long long)blockIdx.x * 256 + threadIdx.x;
  if (i < n) p[i] = v;
}
__global__ void k_count(const int* __restrict__ b, float* g, int n) {
  int i = blockIdx.x * 256 + threadIdx.x;
  if (i < n) atomicAdd(&g[b[i]], 1.0f);
}
__global__ void k_deg(const int* __restrict__ dst, float* deg, int e) {
  int i = blockIdx.x * 256 + threadIdx.x;
  if (i < e) atomicAdd(&deg[dst[i]], 1.0f);
}
__global__ void k_rsqrt(float* p, int n) {
  int i = blockIdx.x * 256 + threadIdx.x;
  if (i < n) p[i] = rsqrtf(p[i]);
}

// ---------------- GCN ------------------------------------------------------
__global__ void k_gcn_edge(const int* __restrict__ src, const int* __restrict__ dst,
                           const float* __restrict__ h, const float* __restrict__ dis,
                           float* __restrict__ agg, int e)
{
  int ed = blockIdx.x * 8 + (threadIdx.x >> 5);
  if (ed >= e) return;
  int lane = threadIdx.x & 31;
  int s = src[ed], d = dst[ed];
  float w = dis[s] * dis[d];
  float4 hv = *(const float4*)(h + (long)s * FF + lane * 4);
  float* ap = agg + (long)d * FF + lane * 4;
  atomicAdd(ap + 0, hv.x * w); atomicAdd(ap + 1, hv.y * w);
  atomicAdd(ap + 2, hv.z * w); atomicAdd(ap + 3, hv.w * w);
}
__global__ void k_gcn_finish(const float* __restrict__ agg, const float* __restrict__ h,
                             const float* __restrict__ dis, const float* __restrict__ b0,
                             float* __restrict__ xout, int n)
{
  long i = (long)blockIdx.x * 256 + threadIdx.x;
  if (i >= (long)n * FF) return;
  int node = i >> 7, f = i & 127;
  float d = dis[node];
  xout[i] = agg[i] + h[i] * d * d + b0[f];       // self loop + bias
}

// ---------------- GraphNorm ------------------------------------------------
__global__ void k_segsum(const float* __restrict__ x, const int* __restrict__ batch,
                         float* __restrict__ gsum, int n)
{
  int f = threadIdx.x & 127, half = threadIdx.x >> 7;
  int base = blockIdx.x * 128;
  int end = base + 128; if (end > n) end = n;
  float local = 0.f; int curg = -1;
  for (int i = base + half; i < end; i += 2) {
    int g = batch[i];
    if (g != curg) { if (curg >= 0) atomicAdd(&gsum[curg * FF + f], local); local = 0.f; curg = g; }
    local += x[(long)i * FF + f];
  }
  if (curg >= 0) atomicAdd(&gsum[curg * FF + f], local);
}
__global__ void k_segsq(const float* __restrict__ x, const int* __restrict__ batch,
                        const float* __restrict__ mean, const float* __restrict__ alpha,
                        float* __restrict__ gsq, int n)
{
  int f = threadIdx.x & 127, half = threadIdx.x >> 7;
  int base = blockIdx.x * 128;
  int end = base + 128; if (end > n) end = n;
  float am = alpha[f];
  float local = 0.f, mg = 0.f; int curg = -1;
  for (int i = base + half; i < end; i += 2) {
    int g = batch[i];
    if (g != curg) {
      if (curg >= 0) atomicAdd(&gsq[curg * FF + f], local);
      local = 0.f; curg = g; mg = mean[g * FF + f];
    }
    float xc = x[(long)i * FF + f] - am * mg;
    local += xc * xc;
  }
  if (curg >= 0) atomicAdd(&gsq[curg * FF + f], local);
}
__global__ void k_div(float* buf, const float* __restrict__ gcnt, int n) {
  int i = blockIdx.x * 256 + threadIdx.x;
  if (i < n) { float c = gcnt[i >> 7]; c = c > 1.f ? c : 1.f; buf[i] /= c; }
}
__global__ void k_gn_apply(const float* __restrict__ x, const int* __restrict__ batch,
                           const float* __restrict__ mean, const float* __restrict__ var,
                           const float* __restrict__ gamma, const float* __restrict__ beta,
                           const float* __restrict__ alpha, const float* __restrict__ prev,
                           float* __restrict__ out, int n)
{
  long i = (long)blockIdx.x * 256 + threadIdx.x;
  if (i >= (long)n * FF) return;
  int node = i >> 7, f = i & 127;
  int g = batch[node];
  float xc = x[i] - alpha[f] * mean[g * FF + f];
  float val = gamma[f] * xc * rsqrtf(var[g * FF + f] + 1e-5f) + beta[f];
  out[i] = prev ? prev[i] + val : val;
}

// ---------------- GAT ------------------------------------------------------
__global__ void k_gat_node(const float* __restrict__ h, const float* __restrict__ atts,
                           const float* __restrict__ attd, float* __restrict__ a_s,
                           float* __restrict__ a_d, unsigned* __restrict__ emax, int n)
{
  int idx = blockIdx.x * 256 + threadIdx.x;
  if (idx >= n * HH) return;
  int node = idx >> 2, hd = idx & 3;
  const float* hp = h + (long)node * FF + hd * CCH;
  float s = 0.f, d = 0.f;
  #pragma unroll
  for (int c = 0; c < CCH; ++c) { float v = hp[c]; s += v * atts[hd * CCH + c]; d += v * attd[hd * CCH + c]; }
  a_s[idx] = s; a_d[idx] = d;
  emax[idx] = encf(leaky(s + d));                  // init with self-loop score
}
__global__ void k_gat_edge_max(const int* __restrict__ src, const int* __restrict__ dst,
                               const float* __restrict__ a_s, const float* __restrict__ a_d,
                               unsigned* __restrict__ emax, int e)
{
  int idx = blockIdx.x * 256 + threadIdx.x;
  if (idx >= e * HH) return;
  int ed = idx >> 2, hd = idx & 3;
  int s = src[ed], d = dst[ed];
  atomicMax(&emax[d * HH + hd], encf(leaky(a_s[s * HH + hd] + a_d[d * HH + hd])));
}
__global__ void k_gat_den_init(const float* __restrict__ a_s, const float* __restrict__ a_d,
                               const unsigned* __restrict__ emax, float* __restrict__ den, int n)
{
  int idx = blockIdx.x * 256 + threadIdx.x;
  if (idx >= n * HH) return;
  den[idx] = __expf(leaky(a_s[idx] + a_d[idx]) - decf(emax[idx]));
}
__global__ void k_gat_edge_den(const int* __restrict__ src, const int* __restrict__ dst,
                               const float* __restrict__ a_s, const float* __restrict__ a_d,
                               const unsigned* __restrict__ emax, float* __restrict__ den, int e)
{
  int idx = blockIdx.x * 256 + threadIdx.x;
  if (idx >= e * HH) return;
  int ed = idx >> 2, hd = idx & 3;
  int s = src[ed], d = dst[ed];
  int ih = d * HH + hd;
  atomicAdd(&den[ih], __expf(leaky(a_s[s * HH + hd] + a_d[ih]) - decf(emax[ih])));
}
__global__ void k_gat_agg_init(const float* __restrict__ h, const float* __restrict__ a_s,
                               const float* __restrict__ a_d, const unsigned* __restrict__ emax,
                               const float* __restrict__ den, float* __restrict__ agg, int n)
{
  long i = (long)blockIdx.x * 256 + threadIdx.x;
  if (i >= (long)n * FF) return;
  int node = i >> 7, hd = (i & 127) >> 5;
  int ih = node * HH + hd;
  float al = __expf(leaky(a_s[ih] + a_d[ih]) - decf(emax[ih])) / (den[ih] + 1e-16f);
  agg[i] = h[i] * al;                              // self-loop contribution
}
__global__ void k_gat_edge_agg(const int* __restrict__ src, const int* __restrict__ dst,
                               const float* __restrict__ h, const float* __restrict__ a_s,
                               const float* __restrict__ a_d, const unsigned* __restrict__ emax,
                               const float* __restrict__ den, float* __restrict__ agg, int e)
{
  int ed = blockIdx.x * 8 + (threadIdx.x >> 5);
  if (ed >= e) return;
  int lane = threadIdx.x & 31;
  int s = src[ed], d = dst[ed];
  int hd = lane >> 3;
  int ih = d * HH + hd;
  float al = __expf(leaky(a_s[s * HH + hd] + a_d[ih]) - decf(emax[ih])) / (den[ih] + 1e-16f);
  float4 hv = *(const float4*)(h + (long)s * FF + lane * 4);
  float* ap = agg + (long)d * FF + lane * 4;
  atomicAdd(ap + 0, hv.x * al); atomicAdd(ap + 1, hv.y * al);
  atomicAdd(ap + 2, hv.z * al); atomicAdd(ap + 3, hv.w * al);
}
__global__ void k_gelu_bias(float* agg, const float* __restrict__ b, int n) {
  long i = (long)blockIdx.x * 256 + threadIdx.x;
  if (i >= (long)n * FF) return;
  float u = agg[i] + b[i & 127];
  agg[i] = 0.5f * u * (1.0f + erff(u * 0.70710678118654752f));
}

// ---------------- host-side orchestration ----------------------------------
struct Ws {
  float *h, *agg, *x, *dis, *as_, *ad_, *den, *gsum, *gsq, *gcnt;
  unsigned* emax;
};
static inline int nb(long long n) { return (int)((n + 255) / 256); }

static void graphnorm(const float* xin, const int* batch, const float* gamma,
                      const float* beta, const float* alpha, const float* prev,
                      float* out, const Ws& w, hipStream_t st)
{
  k_fill<<<nb(GGR * FF), 256, 0, st>>>(w.gsum, 0.f, GGR * FF);
  k_segsum<<<(NN + 127) / 128, 256, 0, st>>>(xin, batch, w.gsum, NN);
  k_div<<<nb(GGR * FF), 256, 0, st>>>(w.gsum, w.gcnt, GGR * FF);      // -> mean
  k_fill<<<nb(GGR * FF), 256, 0, st>>>(w.gsq, 0.f, GGR * FF);
  k_segsq<<<(NN + 127) / 128, 256, 0, st>>>(xin, batch, w.gsum, alpha, w.gsq, NN);
  k_div<<<nb(GGR * FF), 256, 0, st>>>(w.gsq, w.gcnt, GGR * FF);       // -> var
  k_gn_apply<<<nb((long long)NN * FF), 256, 0, st>>>(xin, batch, w.gsum, w.gsq,
                                                     gamma, beta, alpha, prev, out, NN);
}

static void run_branch(const float* xin, const int* src, const int* dst, const int* batch,
                       const float* W0, const float* b0, const float* g0g, const float* g0b,
                       const float* g0a, const float* gatW, const float* attS,
                       const float* attD, const float* gatB, const float* gng,
                       const float* gnb, const float* gna, const float* linW,
                       const float* linb, float* out, const Ws& w, hipStream_t st)
{
  const long long NF = (long long)NN * FF;
  // per-graph node counts
  k_fill<<<1, 256, 0, st>>>(w.gcnt, 0.f, GGR);
  k_count<<<nb(NN), 256, 0, st>>>(batch, w.gcnt, NN);
  // degrees (self loop => init 1), dis = deg^-1/2
  k_fill<<<nb(NN), 256, 0, st>>>(w.dis, 1.f, NN);
  k_deg<<<nb(EE), 256, 0, st>>>(dst, w.dis, EE);
  k_rsqrt<<<nb(NN), 256, 0, st>>>(w.dis, NN);
  // ---- GCNConv ----
  k_gemm<<<(NN + 127) / 128, 256, 0, st>>>(xin, W0, nullptr, w.h, NN, FF);
  k_fill<<<nb(NF), 256, 0, st>>>(w.agg, 0.f, NF);
  k_gcn_edge<<<(EE + 7) / 8, 256, 0, st>>>(src, dst, w.h, w.dis, w.agg, EE);
  k_gcn_finish<<<nb(NF), 256, 0, st>>>(w.agg, w.h, w.dis, b0, w.x, NN);
  graphnorm(w.x, batch, g0g, g0b, g0a, nullptr, w.x, w, st);
  // ---- 2x GATConv + GELU + GraphNorm residual ----
  for (int l = 0; l < 2; ++l) {
    const float* Wl = gatW + (long)l * FF * FF;
    k_gemm<<<(NN + 127) / 128, 256, 0, st>>>(w.x, Wl, nullptr, w.h, NN, FF);
    k_gat_node<<<nb((long long)NN * HH), 256, 0, st>>>(w.h, attS + l * FF, attD + l * FF,
                                                       w.as_, w.ad_, w.emax, NN);
    k_gat_edge_max<<<nb((long long)EE * HH), 256, 0, st>>>(src, dst, w.as_, w.ad_, w.emax, EE);
    k_gat_den_init<<<nb((long long)NN * HH), 256, 0, st>>>(w.as_, w.ad_, w.emax, w.den, NN);
    k_gat_edge_den<<<nb((long long)EE * HH), 256, 0, st>>>(src, dst, w.as_, w.ad_, w.emax, w.den, EE);
    k_gat_agg_init<<<nb(NF), 256, 0, st>>>(w.h, w.as_, w.ad_, w.emax, w.den, w.agg, NN);
    k_gat_edge_agg<<<(EE + 7) / 8, 256, 0, st>>>(src, dst, w.h, w.as_, w.ad_, w.emax, w.den, w.agg, EE);
    k_gelu_bias<<<nb(NF), 256, 0, st>>>(w.agg, gatB + l * FF, NN);
    graphnorm(w.agg, batch, gng + l * FF, gnb + l * FF, gna + l * FF, w.x, w.x, w, st);
  }
  // ---- final linear ----
  k_gemm<<<(NN + 127) / 128, 256, 0, st>>>(w.x, linW, linb, out, NN, FOUT);
}

extern "C" void kernel_launch(void* const* d_in, const int* in_sizes, int n_in,
                              void* d_out, int out_size, void* d_ws, size_t ws_size,
                              hipStream_t stream)
{
  (void)in_sizes; (void)n_in; (void)out_size; (void)ws_size;
  const float* x1 = (const float*)d_in[0];
  const float* x2 = (const float*)d_in[1];
  const int* ei1 = (const int*)d_in[2];
  const int* ei2 = (const int*)d_in[3];
  const int* batch1 = (const int*)d_in[4];
  const int* batch2 = (const int*)d_in[5];
  const float* W0 = (const float*)d_in[6];
  const float* b0 = (const float*)d_in[7];
  const float* g0g = (const float*)d_in[8];
  const float* g0b = (const float*)d_in[9];
  const float* g0a = (const float*)d_in[10];
  const float* gatW = (const float*)d_in[11];
  const float* attS = (const float*)d_in[12];
  const float* attD = (const float*)d_in[13];
  const float* gatB = (const float*)d_in[14];
  const float* gng = (const float*)d_in[15];
  const float* gnb = (const float*)d_in[16];
  const float* gna = (const float*)d_in[17];
  const float* linW = (const float*)d_in[18];
  const float* linb = (const float*)d_in[19];
  float* out = (float*)d_out;

  // workspace partition (~161 MB)
  char* p = (char*)d_ws;
  auto takeF = [&](long long n) { float* r = (float*)p; p += ((n * 4 + 255) & ~255LL); return r; };
  Ws w;
  w.h    = takeF((long long)NN * FF);
  w.agg  = takeF((long long)NN * FF);
  w.x    = takeF((long long)NN * FF);
  w.dis  = takeF(NN);
  w.as_  = takeF((long long)NN * HH);
  w.ad_  = takeF((long long)NN * HH);
  w.den  = takeF((long long)NN * HH);
  w.emax = (unsigned*)takeF((long long)NN * HH);
  w.gsum = takeF(GGR * FF);
  w.gsq  = takeF(GGR * FF);
  w.gcnt = takeF(GGR);

  run_branch(x1, ei1, ei1 + EE, batch1, W0, b0, g0g, g0b, g0a,
             gatW, attS, attD, gatB, gng, gnb, gna, linW, linb,
             out, w, stream);
  run_branch(x2, ei2, ei2 + EE, batch2, W0, b0, g0g, g0b, g0a,
             gatW, attS, attD, gatB, gng, gnb, gna, linW, linb,
             out + (long long)NN * FOUT, w, stream);
}
